// Transformer_9380208575339
// MI455X (gfx1250) — matrix operand order repaired and checked by reading, not compiled, not run on hardware
//
#include <hip/hip_runtime.h>
#include <hip/hip_bf16.h>
#include <math.h>

// GPT-2-ish forward for MI455X (gfx1250, wave32, WMMA + TDM).
// All GEMMs (QKV, QK^T, AV, attn-proj, FC+GELU, proj) run on
// v_wmma_f32_16x16x32_f16 with fp32 accumulation; the main GEMM's A-tile is
// staged into LDS by the Tensor Data Mover (tensor_load_to_lds).

typedef _Float16 f16;
typedef __attribute__((ext_vector_type(16))) _Float16 v16h;
typedef __attribute__((ext_vector_type(8)))  _Float16 v8h;
typedef __attribute__((ext_vector_type(8)))  float    v8f;
typedef __attribute__((ext_vector_type(4)))  unsigned int u32x4;
typedef __attribute__((ext_vector_type(8)))  int i32x8;
typedef __attribute__((ext_vector_type(4)))  int i32x4;

#define E_   1024
#define S_   1024
#define FF_  4096
#define M_   2048   // B*S
#define HD_  64
#define BH_  32     // B * num_heads
#define NLAYER_ 6

static __device__ __forceinline__ v16h pack16(float4 a, float4 b, float4 c, float4 d) {
  v16h r;
  r[0]=(f16)a.x;  r[1]=(f16)a.y;  r[2]=(f16)a.z;  r[3]=(f16)a.w;
  r[4]=(f16)b.x;  r[5]=(f16)b.y;  r[6]=(f16)b.z;  r[7]=(f16)b.w;
  r[8]=(f16)c.x;  r[9]=(f16)c.y;  r[10]=(f16)c.z; r[11]=(f16)c.w;
  r[12]=(f16)d.x; r[13]=(f16)d.y; r[14]=(f16)d.z; r[15]=(f16)d.w;
  return r;
}

static __device__ __forceinline__ v8f wmma32(v16h a, v16h b, v8f c) {
  // 8 args: (neg_a, A, neg_b, B, c_mod, C, reuse_a, reuse_b)
  return __builtin_amdgcn_wmma_f32_16x16x32_f16(false, a, false, b, (short)0, c,
                                                false, false);
}

static __device__ __forceinline__ v16h cat8(v8h lo, v8h hi) {
  return __builtin_shufflevector(lo, hi, 0,1,2,3,4,5,6,7,8,9,10,11,12,13,14,15);
}

// ---------------------------------------------------------------------------
// C[M,N] = act(A[M,K] @ W[K,N] + bias[N])
// Block = 256 threads (8 waves), output tile 128 x 64.
// Per K-step(32):
//   - A-tile (128x32 fp32, stride K) DMA'd to LDS by the TDM (wave 0 issues
//     tensor_load_to_lds, waits TENSORcnt, barrier releases the block)
//   - W-tile (32x64) cooperatively staged transposed as f16
//   - each wave runs 4 independent back-to-back WMMAs (16x64 strip)
// ---------------------------------------------------------------------------
template<int ACT>
__global__ __launch_bounds__(256) void gemm_kernel(const float* __restrict__ A,
                                                   const float* __restrict__ W,
                                                   const float* __restrict__ bias,
                                                   float* __restrict__ C,
                                                   int K, int N) {
  __shared__ __align__(16) float sAf[128 * 32]; // [row][k] fp32 (TDM dest), 16 KB
  __shared__ __align__(16) f16   sW[64 * 40];   // [col][k] f16, pad stride 40
  const int n0   = blockIdx.x * 64;
  const int m0   = blockIdx.y * 128;
  const int tid  = threadIdx.x;
  const int wave = tid >> 5;
  const int lane = tid & 31;
  const int hi   = (lane < 16) ? 0 : 1;        // half-wave selector (ISA layout)
  const int l15  = lane & 15;

  const unsigned int ldsA = (unsigned int)(uintptr_t)(void*)sAf;  // LDS offset

  v8f acc[4] = {v8f{}, v8f{}, v8f{}, v8f{}};

  for (int k0 = 0; k0 < K; k0 += 32) {
    __syncthreads();
    // ---- stage W[k0..k0+31, n0..n0+63] -> sW transposed (f16) ----
#pragma unroll
    for (int i = 0; i < 2; ++i) {
      int g   = tid + i * 256;              // 512 float4 groups
      int row = g >> 4;                     // k: 0..31
      int cq  = (g & 15) * 4;               // col: 0..60
      float4 f = *(const float4*)(W + (size_t)(k0 + row) * N + n0 + cq);
      sW[(cq + 0) * 40 + row] = (f16)f.x;
      sW[(cq + 1) * 40 + row] = (f16)f.y;
      sW[(cq + 2) * 40 + row] = (f16)f.z;
      sW[(cq + 3) * 40 + row] = (f16)f.w;
    }
    // prefetch next W K-slice into cache (global_prefetch_b8)
    if (k0 + 32 < K)
      __builtin_prefetch(W + (size_t)(k0 + 32 + (tid >> 3)) * N + n0 + (tid & 7) * 8, 0, 3);

    // ---- TDM: A[m0..m0+127, k0..k0+31] fp32 -> sAf  (wave 0 issues) ----
    if (wave == 0) {
      unsigned long long ga =
          (unsigned long long)(uintptr_t)(A + (size_t)m0 * K + k0);
      u32x4 g0;
      g0[0] = 1u;                                   // count=1, load descriptor
      g0[1] = ldsA;                                 // lds_addr
      g0[2] = (unsigned int)ga;                     // global_addr[31:0]
      g0[3] = (unsigned int)((ga >> 32) & 0x1FFFFFFu) | (2u << 30); // addr[56:32], type=2
      i32x8 g1;
      g1[0] = 0x20000;                              // data_size=2 (4 bytes)
      g1[1] = (int)((unsigned int)K << 16);         // tensor_dim0 = K (row len)
      g1[2] = (int)((unsigned int)M_ << 16);        // tensor_dim1 = M
      g1[3] = (int)(32u << 16);                     // tile_dim0 = 32
      g1[4] = 128;                                  // tile_dim1 = 128 rows
      g1[5] = K;                                    // tensor_dim0_stride = K
      g1[6] = 0;
      g1[7] = 0;
      i32x4 g2 = {0, 0, 0, 0};
      i32x4 g3 = {0, 0, 0, 0};
      i32x8 g4 = {0, 0, 0, 0, 0, 0, 0, 0};
      __builtin_amdgcn_tensor_load_to_lds(g0, g1, g2, g3, g4, 0);
      __builtin_amdgcn_s_wait_tensorcnt(0);
    }
    __syncthreads();

    // ---- A fragment: wave's 16 rows, fp32 LDS -> f16 regs (ISA A layout) ----
    const float4* ap = (const float4*)(sAf + (wave * 16 + l15) * 32 + hi * 8);
    v16h af = pack16(ap[0], ap[1], ap[4], ap[5]); // k: [hi*8..+7], [hi*8+16..+23]

    // ---- 4 B fragments / 4 WMMAs (16x64 strip) ----
#pragma unroll
    for (int nt = 0; nt < 4; ++nt) {
      const v8h* bp = (const v8h*)(sW + (nt * 16 + l15) * 40 + hi * 8);
      v16h bf = cat8(bp[0], bp[2]);
      acc[nt] = wmma32(af, bf, acc[nt]);
    }
  }

  // ---- epilogue: bias (+GELU), fp32 store ----
#pragma unroll
  for (int nt = 0; nt < 4; ++nt) {
    const float bcol = bias[n0 + nt * 16 + l15];
    float* cp = C + (size_t)(m0 + wave * 16 + hi * 8) * N + n0 + nt * 16 + l15;
#pragma unroll
    for (int r = 0; r < 8; ++r) {
      float v = acc[nt][r] + bcol;
      if (ACT == 1) v = 0.5f * v * (1.0f + erff(v * 0.70710678118654752f));
      cp[(size_t)r * N] = v;
    }
  }
}

// ---------------------------------------------------------------------------
// scores[bh, s, t] = (q . k) / 8 for causal tiles only (tT <= sT), f16 out.
// One wave per 16x16 tile; K = head_dim = 64 -> two WMMA steps.
// ---------------------------------------------------------------------------
__global__ __launch_bounds__(32) void qk_kernel(const float* __restrict__ qkv,
                                                f16* __restrict__ att) {
  const int tT = blockIdx.x, sT = blockIdx.y, bh = blockIdx.z;
  if (tT > sT) return;
  const int b = bh >> 4, n = bh & 15;
  const int lane = threadIdx.x & 31;
  const int hi = (lane < 16) ? 0 : 1;
  const int srow = sT * 16 + (lane & 15);
  const int tcol = tT * 16 + (lane & 15);
  const float* qb = qkv + (size_t)(b * S_ + srow) * (3 * E_) + n * HD_;
  const float* kb = qkv + (size_t)(b * S_ + tcol) * (3 * E_) + E_ + n * HD_;

  v8f acc = {};
#pragma unroll
  for (int k0 = 0; k0 < HD_; k0 += 32) {
    const float4* ap = (const float4*)(qb + k0 + hi * 8);
    v16h af = pack16(ap[0], ap[1], ap[4], ap[5]);
    const float4* bp = (const float4*)(kb + k0 + hi * 8);
    v16h bf = pack16(bp[0], bp[1], bp[4], bp[5]);
    acc = wmma32(af, bf, acc);
  }

  f16* op = att + ((size_t)bh * S_ + sT * 16 + hi * 8) * S_ + tcol;
#pragma unroll
  for (int r = 0; r < 8; ++r) op[(size_t)r * S_] = (f16)(acc[r] * 0.125f);
}

// ---------------------------------------------------------------------------
// Row softmax over the causal prefix; zero-pads the diagonal tile remainder.
// ---------------------------------------------------------------------------
__global__ __launch_bounds__(256) void softmax_kernel(f16* __restrict__ att) {
  const int s = blockIdx.x, bh = blockIdx.y;
  const int tid = threadIdx.x;
  const int limit = ((s >> 4) + 1) * 16;     // tile-padded causal length
  f16* row = att + ((size_t)bh * S_ + s) * S_;
  __shared__ float red[256];

  float mx = -1e30f;
  for (int j = tid; j <= s; j += 256) mx = fmaxf(mx, (float)row[j]);
  red[tid] = mx; __syncthreads();
  for (int o = 128; o; o >>= 1) { if (tid < o) red[tid] = fmaxf(red[tid], red[tid + o]); __syncthreads(); }
  mx = red[0]; __syncthreads();

  float sum = 0.0f;
  for (int j = tid; j <= s; j += 256) sum += __expf((float)row[j] - mx);
  red[tid] = sum; __syncthreads();
  for (int o = 128; o; o >>= 1) { if (tid < o) red[tid] += red[tid + o]; __syncthreads(); }
  const float inv = 1.0f / red[0];

  for (int j = tid; j < limit; j += 256) {
    float v = (j <= s) ? __expf((float)row[j] - mx) * inv : 0.0f;
    row[j] = (f16)v;
  }
}

// ---------------------------------------------------------------------------
// o[bh, s, h] = att[s, :] @ v[:, h]; wave per (hT, sT, bh) 16x16 tile.
// Causal: only iterates t-tiles up to the diagonal.
// ---------------------------------------------------------------------------
__global__ __launch_bounds__(32) void av_kernel(const f16* __restrict__ att,
                                                const float* __restrict__ qkv,
                                                float* __restrict__ obuf) {
  const int hT = blockIdx.x, sT = blockIdx.y, bh = blockIdx.z;
  const int b = bh >> 4, n = bh & 15;
  const int lane = threadIdx.x & 31;
  const int hi = (lane < 16) ? 0 : 1;
  const int limit = (sT + 1) * 16;
  const int srow = sT * 16 + (lane & 15);
  const int hcol = hT * 16 + (lane & 15);
  const f16* arow = att + ((size_t)bh * S_ + srow) * S_;
  const float* vb = qkv + (size_t)b * S_ * 3 * E_ + 2 * E_ + n * HD_ + hcol;

  v8f acc = {};
  for (int k0 = 0; k0 < limit; k0 += 32) {
    v16h af;
    const int kb = k0 + hi * 8;
#pragma unroll
    for (int e = 0; e < 8; ++e) {
      int t0 = kb + e, t1 = kb + 16 + e;
      af[e]     = (t0 < limit) ? arow[t0] : (f16)0.0f;
      af[e + 8] = (t1 < limit) ? arow[t1] : (f16)0.0f;
    }
    v16h bf;
    const int kv = k0 + hi * 8;
#pragma unroll
    for (int e = 0; e < 16; ++e) bf[e] = (f16)vb[(size_t)(kv + e + (e & 8)) * (3 * E_)];
    acc = wmma32(af, bf, acc);
  }

  float* op = obuf + (size_t)(b * S_ + sT * 16 + hi * 8) * E_ + n * HD_ + hcol;
#pragma unroll
  for (int r = 0; r < 8; ++r) op[(size_t)r * E_] = acc[r];
}

// ---------------------------------------------------------------------------
// tmp = wte[tok] + wpe  (elementwise)
// ---------------------------------------------------------------------------
__global__ __launch_bounds__(256) void embed_kernel(const int* __restrict__ tokens,
                                                    const float* __restrict__ wte,
                                                    const float* __restrict__ wpe,
                                                    float* __restrict__ out) {
  int idx = blockIdx.x * 256 + threadIdx.x;
  int e   = idx & (E_ - 1);
  int row = idx >> 10;           // b*S + s
  int s   = row & (S_ - 1);
  int tok = tokens[row];
  out[idx] = wte[(size_t)tok * E_ + e] + wpe[(size_t)s * E_ + e];
}

// ---------------------------------------------------------------------------
// out = LN(in + res) * w + b   (res == nullptr -> plain LN). One block / row.
// ---------------------------------------------------------------------------
__global__ __launch_bounds__(256) void ln_kernel(const float* __restrict__ in,
                                                 const float* __restrict__ res,
                                                 const float* __restrict__ w,
                                                 const float* __restrict__ bias,
                                                 float* __restrict__ out) {
  const int row = blockIdx.x, tid = threadIdx.x;
  const float* ip = in + (size_t)row * E_;
  float v[4]; float s = 0.0f;
#pragma unroll
  for (int i = 0; i < 4; ++i) {
    float t = ip[tid + i * 256];
    if (res) t += res[(size_t)row * E_ + tid + i * 256];
    v[i] = t; s += t;
  }
  __shared__ float red[256];
  red[tid] = s; __syncthreads();
  for (int o = 128; o; o >>= 1) { if (tid < o) red[tid] += red[tid + o]; __syncthreads(); }
  const float mu = red[0] * (1.0f / E_);
  __syncthreads();
  float s2 = 0.0f;
#pragma unroll
  for (int i = 0; i < 4; ++i) { float d = v[i] - mu; s2 += d * d; }
  red[tid] = s2; __syncthreads();
  for (int o = 128; o; o >>= 1) { if (tid < o) red[tid] += red[tid + o]; __syncthreads(); }
  const float rstd = rsqrtf(red[0] * (1.0f / E_) + 1e-5f);
#pragma unroll
  for (int i = 0; i < 4; ++i) {
    int c = tid + i * 256;
    out[(size_t)row * E_ + c] = (v[i] - mu) * rstd * w[c] + bias[c];
  }
}

// ---------------------------------------------------------------------------
extern "C" void kernel_launch(void* const* d_in, const int* in_sizes, int n_in,
                              void* d_out, int out_size, void* d_ws, size_t ws_size,
                              hipStream_t stream) {
  const int*   tokens = (const int*)d_in[0];
  const float* wte   = (const float*)d_in[1];
  const float* wpe   = (const float*)d_in[2];
  const float* ca_w  = (const float*)d_in[3];
  const float* ca_b  = (const float*)d_in[4];
  const float* ap_w  = (const float*)d_in[5];
  const float* ap_b  = (const float*)d_in[6];
  const float* ln1_w = (const float*)d_in[7];
  const float* ln1_b = (const float*)d_in[8];
  const float* fc_w  = (const float*)d_in[9];
  const float* fc_b  = (const float*)d_in[10];
  const float* pj_w  = (const float*)d_in[11];
  const float* pj_b  = (const float*)d_in[12];
  const float* ln2_w = (const float*)d_in[13];
  const float* ln2_b = (const float*)d_in[14];
  const float* lnd_w = (const float*)d_in[15];
  const float* lnd_b = (const float*)d_in[16];
  const float* lnf_w = (const float*)d_in[17];
  const float* lnf_b = (const float*)d_in[18];

  // Workspace layout (~144 MB): x, tmp, qkv, h (fp32) + att (f16, causal use)
  float* x    = (float*)d_ws;
  float* tmp  = x    + (size_t)M_ * E_;
  float* qkv  = tmp  + (size_t)M_ * E_;
  float* hbuf = qkv  + (size_t)M_ * 3 * E_;
  float* obuf = hbuf + (size_t)M_ * FF_;
  f16*   att  = (f16*)(obuf + (size_t)M_ * E_);

  dim3 b256(256), b32(32);

  embed_kernel<<<dim3(M_ * E_ / 256), b256, 0, stream>>>(tokens, wte, wpe, tmp);
  ln_kernel<<<dim3(M_), b256, 0, stream>>>(tmp, nullptr, lnd_w, lnd_b, x);

  for (int l = 0; l < NLAYER_; ++l) {
    gemm_kernel<0><<<dim3(3 * E_ / 64, M_ / 128), b256, 0, stream>>>(
        x, ca_w + (size_t)l * E_ * 3 * E_, ca_b + (size_t)l * 3 * E_, qkv, E_, 3 * E_);
    qk_kernel<<<dim3(64, 64, BH_), b32, 0, stream>>>(qkv, att);
    softmax_kernel<<<dim3(S_, BH_), b256, 0, stream>>>(att);
    av_kernel<<<dim3(HD_ / 16, 64, BH_), b32, 0, stream>>>(att, qkv, obuf);
    gemm_kernel<0><<<dim3(E_ / 64, M_ / 128), b256, 0, stream>>>(
        obuf, ap_w + (size_t)l * E_ * E_, ap_b + (size_t)l * E_, tmp, E_, E_);
    ln_kernel<<<dim3(M_), b256, 0, stream>>>(
        tmp, x, ln1_w + (size_t)l * E_, ln1_b + (size_t)l * E_, x);
    gemm_kernel<1><<<dim3(FF_ / 64, M_ / 128), b256, 0, stream>>>(
        x, fc_w + (size_t)l * E_ * FF_, fc_b + (size_t)l * FF_, hbuf, E_, FF_);
    gemm_kernel<0><<<dim3(E_ / 64, M_ / 128), b256, 0, stream>>>(
        hbuf, pj_w + (size_t)l * FF_ * E_, pj_b + (size_t)l * E_, tmp, FF_, E_);
    ln_kernel<<<dim3(M_), b256, 0, stream>>>(
        tmp, x, ln2_w + (size_t)l * E_, ln2_b + (size_t)l * E_, x);
  }

  ln_kernel<<<dim3(M_), b256, 0, stream>>>(x, nullptr, lnf_w, lnf_b, (float*)d_out);
}
